// DeepSpeedMoEBlock_146028888422
// MI455X (gfx1250) — matrix operand order repaired and checked by reading, not compiled, not run on hardware
//
#include <hip/hip_runtime.h>
#include <hip/hip_bf16.h>

// ---------------------------------------------------------------------------
// MoE block (top-2, capacity-dropped) + shared expert, bf16 WMMA GEMMs.
// Shapes fixed by reference: d=512, E=16, I=384, K(top)=2.
// GEMM: 64x128 block tile, 8 waves, 2x2 v_wmma_f32_16x16x32_bf16 per wave,
// LDS double-buffered, A tile staged with GLOBAL_LOAD_ASYNC_TO_LDS_B128.
// ---------------------------------------------------------------------------

#define ALPHA 1.702f
#define LIMIT 7.0f
#define DMODEL 512
#define NEXP 16
#define IDIM 384
#define N1 (2 * IDIM)   // 768

typedef __bf16 v16bf __attribute__((ext_vector_type(16)));
typedef float  v8f   __attribute__((ext_vector_type(8)));

__device__ __forceinline__ unsigned short f2bf(float f) {
    unsigned u = __float_as_uint(f);
    u += 0x7FFFu + ((u >> 16) & 1u);       // round-to-nearest-even
    return (unsigned short)(u >> 16);
}

// -------------------------------- fp32 -> bf16 ------------------------------
__global__ void k_convert(const float* __restrict__ src,
                          unsigned short* __restrict__ dst, long long n) {
    long long i = (long long)blockIdx.x * 256 + threadIdx.x;
    if (i < n) dst[i] = f2bf(src[i]);
}

// -------------------------------- init --------------------------------------
__global__ void k_init_tok(int* __restrict__ tok_of, int n) {
    int i = blockIdx.x * 256 + threadIdx.x;
    if (i < n) tok_of[i] = -1;
}

// -------------------------------- gating ------------------------------------
// one wave (32 lanes) per token: logits = x . gate_w[e], softmax, top-2
__global__ void k_gating(const float* __restrict__ x,
                         const float* __restrict__ gw,
                         int T,
                         int* __restrict__ idx_s,      // [2T] slot-major
                         float* __restrict__ w_s,      // [2T]
                         float* __restrict__ probs) {  // [T][16]
    const int wave = threadIdx.x >> 5;
    const int lane = threadIdx.x & 31;
    const int t = blockIdx.x * 8 + wave;
    if (t >= T) return;

    float part[NEXP];
#pragma unroll
    for (int e = 0; e < NEXP; ++e) part[e] = 0.0f;

    const float* xr = x + (long long)t * DMODEL;
    for (int kb = 0; kb < DMODEL; kb += 32) {
        float xv = xr[kb + lane];
#pragma unroll
        for (int e = 0; e < NEXP; ++e)
            part[e] += xv * gw[e * DMODEL + kb + lane];
    }
#pragma unroll
    for (int e = 0; e < NEXP; ++e) {
#pragma unroll
        for (int off = 16; off > 0; off >>= 1)
            part[e] += __shfl_xor(part[e], off, 32);
    }
    // softmax (all lanes redundantly)
    float mx = part[0];
#pragma unroll
    for (int e = 1; e < NEXP; ++e) mx = fmaxf(mx, part[e]);
    float p[NEXP]; float s = 0.0f;
#pragma unroll
    for (int e = 0; e < NEXP; ++e) { p[e] = __expf(part[e] - mx); s += p[e]; }
    float inv = 1.0f / s;
#pragma unroll
    for (int e = 0; e < NEXP; ++e) p[e] *= inv;

    // top-2 (lowest index wins ties, matching lax.top_k)
    float b0 = -1.0f; int i0 = 0;
#pragma unroll
    for (int e = 0; e < NEXP; ++e) if (p[e] > b0) { b0 = p[e]; i0 = e; }
    float b1 = -1.0f; int i1 = 0;
#pragma unroll
    for (int e = 0; e < NEXP; ++e) if (e != i0 && p[e] > b1) { b1 = p[e]; i1 = e; }

    if (lane == 0) {
        float wsum = b0 + b1;
        idx_s[t]     = i0;  w_s[t]     = b0 / wsum;
        idx_s[T + t] = i1;  w_s[T + t] = b1 / wsum;
#pragma unroll
        for (int e = 0; e < NEXP; ++e) probs[(long long)t * NEXP + e] = p[e];
    }
}

// ----------------------- capacity assignment (stable) ------------------------
// single block, 1024 threads; slot-major stable rank per expert.
__global__ void k_assign(const int* __restrict__ idx_s,
                         int* __restrict__ pos_s,
                         int* __restrict__ tok_of,
                         int T, int C) {
    __shared__ int wcnt[32][NEXP];
    __shared__ int base[NEXP];
    const int tid = threadIdx.x;
    const int lane = tid & 31, wv = tid >> 5;
    if (tid < NEXP) base[tid] = 0;
    __syncthreads();

    const int slots = 2 * T;
    for (int ch = 0; ch < slots; ch += 1024) {
        const int slot = ch + tid;
        const int e = idx_s[slot];
        int myrank = 0;
#pragma unroll
        for (int ex = 0; ex < NEXP; ++ex) {
            unsigned bal = (unsigned)__ballot(e == ex);
            if (lane == ex) wcnt[wv][ex] = __popc(bal);
            if (e == ex) myrank = __popc(bal & ((1u << lane) - 1u));
        }
        __syncthreads();
        if (tid < NEXP) {
            int b = base[tid];
            for (int w = 0; w < 32; ++w) {
                int c = wcnt[w][tid];
                wcnt[w][tid] = b;
                b += c;
            }
            base[tid] = b;
        }
        __syncthreads();
        const int pos = wcnt[wv][e] + myrank;
        pos_s[slot] = pos;
        if (pos < C) tok_of[e * C + pos] = (slot >= T) ? (slot - T) : slot;
        __syncthreads();
    }
}

// ------------------------------ WMMA GEMM -----------------------------------
// C[m][n] = A[m][:K](bf16) @ W[e][:K][n](bf16) (+bias). Block tile 64x128,
// 8 waves, each wave 32x32 via 2x2 v_wmma_f32_16x16x32_bf16.
// LDS double-buffered; A staged via async global->LDS (ASYNCcnt).
// SWIGLU: out = swiglu(pairwise cols) -> bf16 H[.][N/2]; else fp32 + bias.
template <bool SWIGLU>
__global__ __launch_bounds__(256) void k_gemm(
    const unsigned short* __restrict__ A, long long aPerExpert,
    const int* __restrict__ tokmap, int C_,
    const unsigned short* __restrict__ W,
    const float* __restrict__ bias,
    void* __restrict__ Out, long long outPerExpert,
    int Kdim, int Ndim) {
    const int e  = blockIdx.z;
    const int n0 = blockIdx.x * 128;
    const int m0 = blockIdx.y * 64;
    const int tid = threadIdx.x;
    const int lane = tid & 31, wv = tid >> 5;
    const int waveM = wv >> 2, waveN = wv & 3;   // 2 x 4 wave grid

    // layout: A0 @0 (4KB) | A1 @4096 | B0 @8192 (8KB) | B1 @16384 ; Ct @0 (32KB)
    __shared__ __align__(16) unsigned char smem[32768];
    float* Ct = (float*)smem;
    const unsigned smemBase = (unsigned)(uintptr_t)(void*)smem;  // LDS byte offset

    const unsigned short* Ae = A + (long long)e * aPerExpert;
    const unsigned short* We = W + (long long)e * Kdim * Ndim;
    const float*          be = bias + (long long)e * Ndim;

    // A staging assignment: 64 rows x 4 chunks of 8 bf16 (16B per thread)
    const int arow = tid >> 2, apart = tid & 3;
    long long aRowBase;
    bool aValid = true;
    if (tokmap) {
        int tok = tokmap[(long long)e * C_ + m0 + arow];
        aValid = (tok >= 0);
        aRowBase = (long long)(aValid ? tok : 0) * Kdim;
    } else {
        aRowBase = (long long)(m0 + arow) * Kdim;
    }
    const unsigned aLdsOff = (unsigned)((arow * 32 + apart * 8) * 2);
    // B staging assignment: 16 K-pairs x 16 col-chunks of 8
    const int bj = tid >> 4, bn = tid & 15;

    v8f acc00 = {}, acc01 = {}, acc10 = {}, acc11 = {};
    const int half = lane >> 4, r16 = lane & 15;

    auto stage = [&](int kk, int b) {
        // ---- A tile (64x32): async copy global -> LDS, 16B per lane ----
        if (aValid) {
            const unsigned short* gsrc = Ae + aRowBase + kk + apart * 8;
            unsigned dstA = smemBase + (unsigned)(b * 4096) + aLdsOff;
            asm volatile("global_load_async_to_lds_b128 %0, %1, off"
                         :: "v"(dstA), "v"(gsrc) : "memory");
        } else {
            unsigned short* As = (unsigned short*)(smem + b * 4096);
            *reinterpret_cast<int4*>(As + arow * 32 + apart * 8) =
                make_int4(0, 0, 0, 0);
        }
        // ---- B tile (32x128): interleave K pairs into [128][16] dwords ----
        unsigned* Bs = (unsigned*)(smem + 8192 + b * 8192);
        const unsigned short* wr0 =
            We + (long long)(kk + 2 * bj) * Ndim + n0 + bn * 8;
        union { int4 q; unsigned short u[8]; } r0, r1;
        r0.q = *reinterpret_cast<const int4*>(wr0);
        r1.q = *reinterpret_cast<const int4*>(wr0 + Ndim);
#pragma unroll
        for (int i = 0; i < 8; ++i)
            Bs[(bn * 8 + i) * 16 + bj] =
                (unsigned)r0.u[i] | ((unsigned)r1.u[i] << 16);
    };

    auto compute = [&](int b) {
        const unsigned short* As = (const unsigned short*)(smem + b * 4096);
        const unsigned*       Bs = (const unsigned*)(smem + 8192 + b * 8192);
        union AB { int4 q[2]; v16bf v; };
        AB a0, a1, b0v, b1v;
        const unsigned short* ar0 = As + (waveM * 32 + r16) * 32;
        const unsigned short* ar1 = ar0 + 16 * 32;
        a0.q[0] = *(const int4*)(ar0 + half * 8);
        a0.q[1] = *(const int4*)(ar0 + 16 + half * 8);
        a1.q[0] = *(const int4*)(ar1 + half * 8);
        a1.q[1] = *(const int4*)(ar1 + 16 + half * 8);
        const unsigned* bp0 = Bs + (waveN * 32 + r16) * 16 + half * 4;
        const unsigned* bp1 = bp0 + 16 * 16;
        b0v.q[0] = *(const int4*)(bp0);  b0v.q[1] = *(const int4*)(bp0 + 8);
        b1v.q[0] = *(const int4*)(bp1);  b1v.q[1] = *(const int4*)(bp1 + 8);

        acc00 = __builtin_amdgcn_wmma_f32_16x16x32_bf16(false, a0.v, false, b0v.v,
                                                        (short)0, acc00, false, false);
        acc01 = __builtin_amdgcn_wmma_f32_16x16x32_bf16(false, a0.v, false, b1v.v,
                                                        (short)0, acc01, false, false);
        acc10 = __builtin_amdgcn_wmma_f32_16x16x32_bf16(false, a1.v, false, b0v.v,
                                                        (short)0, acc10, false, false);
        acc11 = __builtin_amdgcn_wmma_f32_16x16x32_bf16(false, a1.v, false, b1v.v,
                                                        (short)0, acc11, false, false);
    };

    // ---- double-buffered main loop ----
    stage(0, 0);
    int buf = 0;
    for (int kk = 0; kk < Kdim; kk += 32) {
        asm volatile("s_wait_asynccnt 0x0" ::: "memory");  // our async A writes
        __syncthreads();                                    // + compiler DS waits
        if (kk + 32 < Kdim) stage(kk + 32, buf ^ 1);
        compute(buf);
        buf ^= 1;
    }
    __syncthreads();
    // ---- spill accumulators to LDS tile (C/D layout: VGPR v -> M=v+8*half) --
    {
        const int rb = waveM * 32 + half * 8;
        const int cb = waveN * 32 + r16;
#pragma unroll
        for (int v = 0; v < 8; ++v) {
            Ct[(rb + v) * 128 + cb]           = acc00[v];
            Ct[(rb + v) * 128 + cb + 16]      = acc01[v];
            Ct[(rb + 16 + v) * 128 + cb]      = acc10[v];
            Ct[(rb + 16 + v) * 128 + cb + 16] = acc11[v];
        }
    }
    __syncthreads();

    if constexpr (SWIGLU) {
        unsigned short* H = (unsigned short*)Out;
        const int g = tid >> 4, l16 = tid & 15;
        const int Nh = Ndim >> 1;
#pragma unroll
        for (int it = 0; it < 4; ++it) {
            const int row = g + it * 16;
            const float4 p0 = *(const float4*)&Ct[row * 128 + l16 * 8];
            const float4 p1 = *(const float4*)&Ct[row * 128 + l16 * 8 + 4];
            const float4 q0 = *(const float4*)&be[n0 + l16 * 8];
            const float4 q1 = *(const float4*)&be[n0 + l16 * 8 + 4];
            float gv[4], lv[4], h[4];
            gv[0] = p0.x + q0.x; lv[0] = p0.y + q0.y;
            gv[1] = p0.z + q0.z; lv[1] = p0.w + q0.w;
            gv[2] = p1.x + q1.x; lv[2] = p1.y + q1.y;
            gv[3] = p1.z + q1.z; lv[3] = p1.w + q1.w;
#pragma unroll
            for (int i = 0; i < 4; ++i) {
                float gg = fminf(fmaxf(gv[i], -LIMIT), LIMIT);
                float ll = fminf(fmaxf(lv[i], -LIMIT), LIMIT);
                float sg = 1.0f / (1.0f + __expf(-ALPHA * gg));
                h[i] = gg * sg + (ll + 1.0f);
            }
            uint2 st;
            st.x = (unsigned)f2bf(h[0]) | ((unsigned)f2bf(h[1]) << 16);
            st.y = (unsigned)f2bf(h[2]) | ((unsigned)f2bf(h[3]) << 16);
            *(uint2*)&H[(long long)e * outPerExpert +
                        (long long)(m0 + row) * Nh + (n0 >> 1) + l16 * 4] = st;
        }
    } else {
        float* O = (float*)Out;
        const int rr = tid >> 5, c4 = (tid & 31) * 4;
#pragma unroll
        for (int it = 0; it < 8; ++it) {
            const int row = rr + it * 8;
            float4 v = *(const float4*)&Ct[row * 128 + c4];
            const float4 bb = *(const float4*)&be[n0 + c4];
            v.x += bb.x; v.y += bb.y; v.z += bb.z; v.w += bb.w;
            *(float4*)&O[(long long)e * outPerExpert +
                         (long long)(m0 + row) * Ndim + n0 + c4] = v;
        }
    }
}

// ------------------------------ combine -------------------------------------
__global__ void k_combine(float* __restrict__ out,
                          const float* __restrict__ out_moe,
                          const int* __restrict__ idx_s,
                          const int* __restrict__ pos_s,
                          const float* __restrict__ w_s,
                          int T, int C) {
    const int t = blockIdx.x;
    const int c = threadIdx.x * 2;
    float2 v = *(float2*)&out[(long long)t * DMODEL + c];
#pragma unroll
    for (int k = 0; k < 2; ++k) {
        const int slot = k * T + t;
        const int e = idx_s[slot];
        const int p = pos_s[slot];
        if (p < C) {
            const float w = w_s[slot];
            const float2 m = *(const float2*)&out_moe[((long long)e * C + p) * DMODEL + c];
            v.x += w * m.x;
            v.y += w * m.y;
        }
    }
    *(float2*)&out[(long long)t * DMODEL + c] = v;
}

// ------------------- deterministic me/ce reductions --------------------------
__global__ void k_reduce(const float* __restrict__ probs,
                         const int* __restrict__ idx_s,
                         float* __restrict__ acc, int T) {
    __shared__ float sm[256], sc[256];
    const int e = blockIdx.x;
    float m = 0.0f, c = 0.0f;
    for (int t = threadIdx.x; t < T; t += 256) {
        m += probs[(long long)t * NEXP + e];
        c += (idx_s[t] == e) ? 1.0f : 0.0f;
    }
    sm[threadIdx.x] = m; sc[threadIdx.x] = c;
    __syncthreads();
    for (int s = 128; s > 0; s >>= 1) {
        if (threadIdx.x < s) {
            sm[threadIdx.x] += sm[threadIdx.x + s];
            sc[threadIdx.x] += sc[threadIdx.x + s];
        }
        __syncthreads();
    }
    if (threadIdx.x == 0) { acc[e] = sm[0]; acc[NEXP + e] = sc[0]; }
}

__global__ void k_finalize(const float* __restrict__ acc,
                           float* __restrict__ out, int T) {
    float s = 0.0f;
    const float invT = 1.0f / (float)T;
#pragma unroll
    for (int e = 0; e < NEXP; ++e)
        s += (acc[e] * invT) * (acc[NEXP + e] * invT);
    out[(long long)T * DMODEL] = (float)NEXP * s;
}

// ---------------------------------------------------------------------------
extern "C" void kernel_launch(void* const* d_in, const int* in_sizes, int n_in,
                              void* d_out, int out_size, void* d_ws, size_t ws_size,
                              hipStream_t stream) {
    const float* x    = (const float*)d_in[0];
    const float* gw   = (const float*)d_in[1];
    const float* W1   = (const float*)d_in[2];
    const float* b1   = (const float*)d_in[3];
    const float* W2   = (const float*)d_in[4];
    const float* b2   = (const float*)d_in[5];
    const float* sw1  = (const float*)d_in[6];
    const float* sb1  = (const float*)d_in[7];
    const float* sw2  = (const float*)d_in[8];
    const float* sb2  = (const float*)d_in[9];
    float* out = (float*)d_out;

    const int T = in_sizes[0] / DMODEL;           // 32768
    const int C = (2 * T) / NEXP;                 // 4096

    // --- workspace carve-out (256B aligned) ---
    char* p = (char*)d_ws;
    auto alloc = [&](size_t bytes) -> char* {
        char* r = p;
        p += (bytes + 255) & ~(size_t)255;
        return r;
    };
    unsigned short* xb16  = (unsigned short*)alloc((size_t)T * DMODEL * 2);
    unsigned short* w1b   = (unsigned short*)alloc((size_t)NEXP * DMODEL * N1 * 2);
    unsigned short* w2b   = (unsigned short*)alloc((size_t)NEXP * IDIM * DMODEL * 2);
    unsigned short* sw1b  = (unsigned short*)alloc((size_t)DMODEL * N1 * 2);
    unsigned short* sw2b  = (unsigned short*)alloc((size_t)IDIM * DMODEL * 2);
    unsigned short* h_e   = (unsigned short*)alloc((size_t)NEXP * C * IDIM * 2);
    unsigned short* h_s   = (unsigned short*)alloc((size_t)T * IDIM * 2);
    float* out_moe        = (float*)alloc((size_t)NEXP * C * DMODEL * 4);
    float* probs          = (float*)alloc((size_t)T * NEXP * 4);
    int*   idx_s          = (int*)alloc((size_t)2 * T * 4);
    int*   pos_s          = (int*)alloc((size_t)2 * T * 4);
    float* w_s            = (float*)alloc((size_t)2 * T * 4);
    int*   tok_of         = (int*)alloc((size_t)NEXP * C * 4);
    float* acc            = (float*)alloc(128);

    auto cvblk = [](long long n) { return (unsigned)((n + 255) / 256); };

    // 1) init + conversions
    k_init_tok<<<cvblk(NEXP * C), 256, 0, stream>>>(tok_of, NEXP * C);
    k_convert<<<cvblk((long long)T * DMODEL), 256, 0, stream>>>(x, xb16, (long long)T * DMODEL);
    k_convert<<<cvblk((long long)NEXP * DMODEL * N1), 256, 0, stream>>>(W1, w1b, (long long)NEXP * DMODEL * N1);
    k_convert<<<cvblk((long long)NEXP * IDIM * DMODEL), 256, 0, stream>>>(W2, w2b, (long long)NEXP * IDIM * DMODEL);
    k_convert<<<cvblk((long long)DMODEL * N1), 256, 0, stream>>>(sw1, sw1b, (long long)DMODEL * N1);
    k_convert<<<cvblk((long long)IDIM * DMODEL), 256, 0, stream>>>(sw2, sw2b, (long long)IDIM * DMODEL);

    // 2) gating (wave per token) + stable capacity assignment
    k_gating<<<dim3(T / 8), 256, 0, stream>>>(x, gw, T, idx_s, w_s, probs);
    k_assign<<<1, 1024, 0, stream>>>(idx_s, pos_s, tok_of, T, C);

    // 3) shared expert GEMM1 (swiglu) : [T,512] @ [512,768] -> h_s [T,384] bf16
    k_gemm<true><<<dim3(N1 / 128, T / 64, 1), 256, 0, stream>>>(
        xb16, 0, nullptr, 0, sw1b, sb1, h_s, 0, DMODEL, N1);
    // 4) expert GEMM1 (gather + swiglu): per expert [C,512] @ [512,768] -> h_e
    k_gemm<true><<<dim3(N1 / 128, C / 64, NEXP), 256, 0, stream>>>(
        xb16, 0, tok_of, C, w1b, b1, h_e, (long long)C * IDIM, DMODEL, N1);
    // 5) shared expert GEMM2 -> d_out (fp32, +sb2)
    k_gemm<false><<<dim3(DMODEL / 128, T / 64, 1), 256, 0, stream>>>(
        h_s, 0, nullptr, 0, sw2b, sb2, out, 0, IDIM, DMODEL);
    // 6) expert GEMM2 -> out_moe (fp32, +b2)
    k_gemm<false><<<dim3(DMODEL / 128, C / 64, NEXP), 256, 0, stream>>>(
        h_e, (long long)C * IDIM, nullptr, 0, w2b, b2, out_moe,
        (long long)C * DMODEL, IDIM, DMODEL);

    // 7) combine expert outputs onto shared output
    k_combine<<<dim3(T), 256, 0, stream>>>(out, out_moe, idx_s, pos_s, w_s, T, C);

    // 8) aux loss (deterministic reductions)
    k_reduce<<<dim3(NEXP), 256, 0, stream>>>(probs, idx_s, acc, T);
    k_finalize<<<1, 1, 0, stream>>>(acc, out, T);
}
